// ContextTransformer_14955076124951
// MI455X (gfx1250) — matrix operand order repaired and checked by reading, not compiled, not run on hardware
//
#include <hip/hip_runtime.h>
#include <hip/hip_bf16.h>

#define NN 50000
#define FF 128
#define KK 16
#define TT 4096

typedef __attribute__((ext_vector_type(16))) __bf16 v16bf;
typedef __attribute__((ext_vector_type(8)))  float  v8f;

union ABfrag { v16bf v; int4 i[2]; };

__device__ __forceinline__ unsigned short f2bf(float f) {
    unsigned u = __float_as_uint(f);
    u += 0x7fffu + ((u >> 16) & 1u);   // round-to-nearest-even
    return (unsigned short)(u >> 16);
}

// ---------------------------------------------------------------------------
// K0: transpose + convert the 5 weight matrices to bf16 (Wt[n][k] = W[k][n])
// grid (64,5) x 256
// ---------------------------------------------------------------------------
__global__ void prep_weights_kernel(const float* Wq, const float* Wk, const float* Wv,
                                    const float* Wo, const float* Wl,
                                    unsigned short* wt) {
    int wsel = blockIdx.y;
    const float* src = wsel == 0 ? Wq : wsel == 1 ? Wk : wsel == 2 ? Wv : wsel == 3 ? Wo : Wl;
    int idx = blockIdx.x * blockDim.x + threadIdx.x;   // 0..16383
    int k = idx >> 7, n = idx & 127;
    wt[wsel * FF * FF + n * FF + k] = f2bf(src[k * FF + n]);
}

// ---------------------------------------------------------------------------
// K1: pre-LayerNorm. 1 wave per row (4 f32/lane). grid 6250 x 256
// outputs: flnF (f32, residual) and flnB (bf16, WMMA A operand)
// ---------------------------------------------------------------------------
__global__ void ln1_kernel(const float* __restrict__ feat,
                           const float* __restrict__ g1, const float* __restrict__ b1,
                           float* __restrict__ flnF, unsigned short* __restrict__ flnB) {
    int row  = blockIdx.x * 8 + (threadIdx.x >> 5);
    int lane = threadIdx.x & 31;
    const float4 x = *(const float4*)(feat + row * FF + lane * 4);
    float s = x.x + x.y + x.z + x.w;
    float q = x.x * x.x + x.y * x.y + x.z * x.z + x.w * x.w;
    #pragma unroll
    for (int m = 1; m < 32; m <<= 1) {
        s += __shfl_xor(s, m, 32);
        q += __shfl_xor(q, m, 32);
    }
    float mean = s * (1.0f / 128.0f);
    float var  = q * (1.0f / 128.0f) - mean * mean;
    float rstd = rsqrtf(var + 1e-5f);
    const float4 g = *(const float4*)(g1 + lane * 4);
    const float4 b = *(const float4*)(b1 + lane * 4);
    float4 y;
    y.x = (x.x - mean) * rstd * g.x + b.x;
    y.y = (x.y - mean) * rstd * g.y + b.y;
    y.z = (x.z - mean) * rstd * g.z + b.z;
    y.w = (x.w - mean) * rstd * g.w + b.w;
    *(float4*)(flnF + row * FF + lane * 4) = y;
    uint2 pk;
    pk.x = (unsigned)f2bf(y.x) | ((unsigned)f2bf(y.y) << 16);
    pk.y = (unsigned)f2bf(y.z) | ((unsigned)f2bf(y.w) << 16);
    *(uint2*)(flnB + row * FF + lane * 4) = pk;
}

// ---------------------------------------------------------------------------
// 16x16 output tile, K=128 reduced in 4 chunks of 32 via v_wmma_f32_16x16x32_bf16
// A: row-major bf16 [rows][128]; Bt: transposed weight bf16 [n][k]
// ---------------------------------------------------------------------------
__device__ __forceinline__ v8f wmma_tile_16x16_k128(const unsigned short* A,
                                                    const unsigned short* Bt,
                                                    int m0, int n0, int lane, v8f c) {
    int mr  = m0 + (lane & 15);
    int col = n0 + (lane & 15);
    int hi  = lane >> 4;
    #pragma unroll
    for (int kk = 0; kk < 4; ++kk) {
        int k0 = kk * 32;
        ABfrag a, b;
        // A (16-bit 16x32 layout): lo lanes k {0..7,16..23}, hi lanes k {8..15,24..31}
        a.i[0] = *(const int4*)(A + mr * FF + k0 + hi * 8);
        a.i[1] = *(const int4*)(A + mr * FF + k0 + 16 + hi * 8);
        // B (32x16): same k order as A: lo lanes k {0..7,16..23}, hi lanes k {8..15,24..31}
        b.i[0] = *(const int4*)(Bt + col * FF + k0 + hi * 8);
        b.i[1] = *(const int4*)(Bt + col * FF + k0 + 16 + hi * 8);
        c = __builtin_amdgcn_wmma_f32_16x16x32_bf16(false, a.v, false, b.v,
                                                    (short)0, c, false, false);
    }
    return c;
}

// ---------------------------------------------------------------------------
// K2: QKV GEMM. grid (3125, 3) x 256; blockIdx.y selects Q/K/V; wave -> col tile
// ---------------------------------------------------------------------------
__global__ void qkv_kernel(const unsigned short* __restrict__ flnB,
                           const unsigned short* __restrict__ wt,
                           const float* __restrict__ bq, const float* __restrict__ bk,
                           const float* __restrict__ bv,
                           float* __restrict__ qo, float* __restrict__ ko,
                           float* __restrict__ vo) {
    int w = threadIdx.x >> 5, lane = threadIdx.x & 31;
    int m0 = blockIdx.x * 16;
    int sel = blockIdx.y;
    const unsigned short* Bt = wt + sel * FF * FF;
    const float* bias = sel == 0 ? bq : (sel == 1 ? bk : bv);
    float* out        = sel == 0 ? qo : (sel == 1 ? ko : vo);
    int n0 = w * 16;
    v8f c = {};
    c = wmma_tile_16x16_k128(flnB, Bt, m0, n0, lane, c);
    int col = n0 + (lane & 15), hi = lane >> 4;
    float bb = bias[col];
    #pragma unroll
    for (int r = 0; r < 8; ++r)
        out[(m0 + r + 8 * hi) * FF + col] = c[r] + bb;
}

// ---------------------------------------------------------------------------
// K3: per-node attention, 1 wave per node (8 nodes / 256-thread block)
// lane l covers features 4l..4l+3 (head = l>>2); grid 6250 x 256
// ---------------------------------------------------------------------------
__global__ void attn_kernel(const float* __restrict__ xyz, const int* __restrict__ edges,
                            const float* __restrict__ qb, const float* __restrict__ kb,
                            const float* __restrict__ vb, const float* __restrict__ table,
                            unsigned short* __restrict__ oB) {
    __shared__ int      sE[8][KK];
    __shared__ unsigned sH[8][KK];
    __shared__ float    sS[8][FF];
    __shared__ float    sA[8][FF];

    int w = threadIdx.x >> 5, lane = threadIdx.x & 31;
    int n = blockIdx.x * 8 + w;

    // stage neighbor ids + spatial hashes
    if (lane < KK) {
        int e = edges[n * KK + lane];
        sE[w][lane] = e;
        float rx = xyz[e * 3 + 0] - xyz[n * 3 + 0];
        float ry = xyz[e * 3 + 1] - xyz[n * 3 + 1];
        float rz = xyz[e * 3 + 2] - xyz[n * 3 + 2];
        unsigned cx = (unsigned)(int)floorf(rx * 64.0f);
        unsigned cy = (unsigned)(int)floorf(ry * 64.0f);
        unsigned cz = (unsigned)(int)floorf(rz * 64.0f);
        unsigned h = (cx * 1u) ^ (cy * 2654435761u) ^ (cz * 805459861u);
        sH[w][lane] = h & (TT - 1);
    }
    const float4 q4 = *(const float4*)(qb + n * FF + lane * 4);
    __syncthreads();

    // scores: lane handles head l>>2, 4 dh-features, one neighbor per iter
    int h = lane >> 2;
    for (int j = 0; j < KK; ++j) {
        int e = sE[w][j];
        unsigned hh = sH[w][j];
        const float4 k4  = *(const float4*)(kb + e * FF + lane * 4);
        const float4 pe4 = *(const float4*)(table + hh * FF + lane * 4);
        float p = q4.x * (k4.x + pe4.x) + q4.y * (k4.y + pe4.y) +
                  q4.z * (k4.z + pe4.z) + q4.w * (k4.w + pe4.w);
        p += __shfl_xor(p, 1, 32);
        p += __shfl_xor(p, 2, 32);   // full dh=16 dot within the 4-lane head group
        if ((lane & 3) == 0) sS[w][h * KK + j] = p * 0.25f;  // 1/sqrt(16)
    }
    __syncthreads();

    // softmax over the 16 neighbors of each head (4 scores/lane, 4 lanes/head)
    {
        int j0 = lane & 3;
        float s0 = sS[w][h * KK + j0 + 0];
        float s1 = sS[w][h * KK + j0 + 4];
        float s2 = sS[w][h * KK + j0 + 8];
        float s3 = sS[w][h * KK + j0 + 12];
        float m = fmaxf(fmaxf(s0, s1), fmaxf(s2, s3));
        m = fmaxf(m, __shfl_xor(m, 1, 32));
        m = fmaxf(m, __shfl_xor(m, 2, 32));
        float e0 = __expf(s0 - m), e1 = __expf(s1 - m);
        float e2 = __expf(s2 - m), e3 = __expf(s3 - m);
        float t = e0 + e1 + e2 + e3;
        t += __shfl_xor(t, 1, 32);
        t += __shfl_xor(t, 2, 32);
        float inv = 1.0f / t;
        sA[w][h * KK + j0 + 0]  = e0 * inv;
        sA[w][h * KK + j0 + 4]  = e1 * inv;
        sA[w][h * KK + j0 + 8]  = e2 * inv;
        sA[w][h * KK + j0 + 12] = e3 * inv;
    }
    __syncthreads();

    // weighted V accumulation
    float4 acc = {0.f, 0.f, 0.f, 0.f};
    for (int j = 0; j < KK; ++j) {
        int e = sE[w][j];
        float a = sA[w][h * KK + j];
        const float4 v4 = *(const float4*)(vb + e * FF + lane * 4);
        acc.x += a * v4.x; acc.y += a * v4.y;
        acc.z += a * v4.z; acc.w += a * v4.w;
    }
    uint2 pk;
    pk.x = (unsigned)f2bf(acc.x) | ((unsigned)f2bf(acc.y) << 16);
    pk.y = (unsigned)f2bf(acc.z) | ((unsigned)f2bf(acc.w) << 16);
    *(uint2*)(oB + n * FF + lane * 4) = pk;
}

// ---------------------------------------------------------------------------
// K4: fused o@Wo + bo + residual -> LN2 -> @Wl + bl + residual. grid 3125 x 256
// Row tile of 16 stays resident in LDS between the two WMMA GEMMs.
// ---------------------------------------------------------------------------
__global__ void out_kernel(const unsigned short* __restrict__ oB,
                           const unsigned short* __restrict__ wt,
                           const float* __restrict__ bo, const float* __restrict__ flnF,
                           const float* __restrict__ g2, const float* __restrict__ b2,
                           const float* __restrict__ bl, float* __restrict__ y) {
    __shared__ float          f2[16][FF];
    __shared__ unsigned short f2b[16][FF];

    int w = threadIdx.x >> 5, lane = threadIdx.x & 31;
    int m0 = blockIdx.x * 16;
    const unsigned short* WoT = wt + 3 * FF * FF;
    const unsigned short* WlT = wt + 4 * FF * FF;

    // GEMM1: attn_out tile = oB @ Wo ; feat2 = fln + attn_out + bo -> LDS (f32)
    v8f c = {};
    c = wmma_tile_16x16_k128(oB, WoT, m0, w * 16, lane, c);
    int col = w * 16 + (lane & 15), hi = lane >> 4;
    float bov = bo[col];
    #pragma unroll
    for (int r = 0; r < 8; ++r) {
        int lr = r + 8 * hi;
        f2[lr][col] = c[r] + bov + flnF[(m0 + lr) * FF + col];
    }
    __syncthreads();

    // LN2 on the resident tile -> bf16 (16 threads/row, 8 elems/thread)
    {
        int row = threadIdx.x >> 4;
        int off = (threadIdx.x & 15) * 8;
        float x[8], s = 0.f, q = 0.f;
        #pragma unroll
        for (int i = 0; i < 8; ++i) { x[i] = f2[row][off + i]; s += x[i]; q += x[i] * x[i]; }
        #pragma unroll
        for (int m = 1; m < 16; m <<= 1) {
            s += __shfl_xor(s, m, 32);
            q += __shfl_xor(q, m, 32);
        }
        float mean = s * (1.0f / 128.0f);
        float var  = q * (1.0f / 128.0f) - mean * mean;
        float rstd = rsqrtf(var + 1e-5f);
        #pragma unroll
        for (int i = 0; i < 8; ++i) {
            int cc = off + i;
            f2b[row][cc] = f2bf((x[i] - mean) * rstd * g2[cc] + b2[cc]);
        }
    }
    __syncthreads();

    // GEMM2: y = LN(feat2) @ Wl + bl + feat2 (A operand streamed from LDS)
    v8f c2 = {};
    c2 = wmma_tile_16x16_k128(&f2b[0][0], WlT, 0, w * 16, lane, c2);
    float blv = bl[col];
    #pragma unroll
    for (int r = 0; r < 8; ++r) {
        int lr = r + 8 * hi;
        y[(m0 + lr) * FF + col] = c2[r] + blv + f2[lr][col];
    }
}

// ---------------------------------------------------------------------------
extern "C" void kernel_launch(void* const* d_in, const int* in_sizes, int n_in,
                              void* d_out, int out_size, void* d_ws, size_t ws_size,
                              hipStream_t stream) {
    const float* xyz   = (const float*)d_in[0];
    const float* feat  = (const float*)d_in[1];
    const int*   edges = (const int*)d_in[2];
    const float* Wq = (const float*)d_in[3];  const float* bq = (const float*)d_in[4];
    const float* Wk = (const float*)d_in[5];  const float* bk = (const float*)d_in[6];
    const float* Wv = (const float*)d_in[7];  const float* bv = (const float*)d_in[8];
    const float* Wo = (const float*)d_in[9];  const float* bo = (const float*)d_in[10];
    const float* table = (const float*)d_in[11];
    const float* g1 = (const float*)d_in[12]; const float* b1 = (const float*)d_in[13];
    const float* g2 = (const float*)d_in[14]; const float* b2 = (const float*)d_in[15];
    const float* Wl = (const float*)d_in[16]; const float* bl = (const float*)d_in[17];
    float* out = (float*)d_out;

    char* w = (char*)d_ws;
    auto alloc = [&](size_t bytes) {
        char* p = w;
        w += (bytes + 255) & ~(size_t)255;
        return p;
    };
    float*          flnF = (float*)alloc((size_t)NN * FF * 4);
    unsigned short* flnB = (unsigned short*)alloc((size_t)NN * FF * 2);
    float*          qb   = (float*)alloc((size_t)NN * FF * 4);
    float*          kbuf = (float*)alloc((size_t)NN * FF * 4);
    float*          vbuf = (float*)alloc((size_t)NN * FF * 4);
    unsigned short* oB   = (unsigned short*)alloc((size_t)NN * FF * 2);
    unsigned short* wt   = (unsigned short*)alloc((size_t)5 * FF * FF * 2);

    prep_weights_kernel<<<dim3(64, 5), 256, 0, stream>>>(Wq, Wk, Wv, Wo, Wl, wt);
    ln1_kernel<<<NN / 8, 256, 0, stream>>>(feat, g1, b1, flnF, flnB);
    qkv_kernel<<<dim3(NN / 16, 3), 256, 0, stream>>>(flnB, wt, bq, bk, bv, qb, kbuf, vbuf);
    attn_kernel<<<NN / 8, 256, 0, stream>>>(xyz, edges, qb, kbuf, vbuf, table, oB);
    out_kernel<<<NN / 16, 256, 0, stream>>>(oB, wt, bo, flnF, g2, b2, bl, out);
}
